// CondensedAttentionNeuralBlock_4741643894863
// MI455X (gfx1250) — matrix operand order repaired and checked by reading, not compiled, not run on hardware
//
#include <hip/hip_runtime.h>
#include <hip/hip_bf16.h>
#include <math.h>

// ---------------------------------------------------------------------------
// CondensedAttentionNeuralBlock for MI455X (gfx1250, wave32, WMMA).
// Latency-bound problem (~2MB tensors, ~3 GFLOP): fuse everything, never
// materialize the 537MB spatial-attention matrix (flash-style online softmax),
// use v_wmma_f32_16x16x32_f16 for the three real GEMMs and the QK^T tiles.
// B tiles are staged in LDS in operand-major (transposed) layout so both A and
// B operand builds vectorize to ds_load_b128; weight panels and activation
// tiles are staged with 128-bit vector moves.
// ---------------------------------------------------------------------------

typedef __attribute__((ext_vector_type(16))) _Float16 v16h;
typedef __attribute__((ext_vector_type(8)))  float    v8f;

__device__ inline v8f zero8() {
    v8f z;
#pragma unroll
    for (int i = 0; i < 8; ++i) z[i] = 0.0f;
    return z;
}
__device__ inline v16h zero16h() {
    v16h z;
#pragma unroll
    for (int i = 0; i < 16; ++i) z[i] = (_Float16)0.0f;
    return z;
}

// CDNA5 16-bit A operand (16x32), row-major source Am[M][ld]:
// lane L: m = L&15 ; K range {ks..ks+7, ks+16..ks+23}, ks = (L<16)?0:8
__device__ inline v16h build_A16(const _Float16* Am, int ld, int mtile, int kk, int lane) {
    v16h a;
    int m  = mtile * 16 + (lane & 15);
    int ks = (lane < 16) ? 0 : 8;
    const _Float16* p = Am + m * ld + kk + ks;
#pragma unroll
    for (int j = 0; j < 8; ++j) a[j] = p[j];          // K = kk+ks+j
#pragma unroll
    for (int j = 8; j < 16; ++j) a[j] = p[j + 8];     // K = kk+ks+16+(j-8)
    return a;
}

// CDNA5 16-bit B operand (32x16) from a TRANSPOSED source Bt[N][ldk] (n-major):
// lane L: n = L&15 ; K range {kb..kb+7, kb+16..kb+23}, kb = (L<16)?0:8 (matches build_A16).
__device__ inline v16h build_BT16(const _Float16* Bt, int ldk, int ntile, int kk, int lane) {
    v16h b;
    int n  = ntile * 16 + (lane & 15);
    int kb = (lane < 16) ? 0 : 8;
    const _Float16* p = Bt + n * ldk + kk + kb;
#pragma unroll
    for (int j = 0; j < 16; ++j) b[j] = p[j < 8 ? j : j + 8];  // two 8-half runs -> 2x ds_load_b128
    return b;
}

__device__ inline v8f wmma_f16(v16h a, v16h b, v8f c) {
    return __builtin_amdgcn_wmma_f32_16x16x32_f16(false, a, false, b, (short)0, c, false, false);
}

// 16-lane-half reductions (C/D rows live across 16 lanes of a wave32 half)
__device__ inline float hred_max(float v) {
#pragma unroll
    for (int m = 8; m >= 1; m >>= 1) v = fmaxf(v, __shfl_xor(v, m, 32));
    return v;
}
__device__ inline float hred_sum(float v) {
#pragma unroll
    for (int m = 8; m >= 1; m >>= 1) v += __shfl_xor(v, m, 32);
    return v;
}
__device__ inline float wred_sum32(float v) {
#pragma unroll
    for (int m = 16; m >= 1; m >>= 1) v += __shfl_xor(v, m, 32);
    return v;
}

// ---------------------------------------------------------------------------
// K0: fold (1x1 conv ∘ depthwise2x2) and (grouped1x1 ∘ pixelshuffle ∘ 1x1)
// into 4+4 effective 64x64 f16 GEMM weights + fused biases.
// ---------------------------------------------------------------------------
__global__ __launch_bounds__(256) void k_prep(
    const float* w_sq1, const float* b_sq1, const float* w_sq2, const float* b_sq2,
    const float* w_un1, const float* b_un1, const float* w_un2, const float* b_un2,
    _Float16* Ae, float* Abias, _Float16* We, float* Wbias) {
    int tid = threadIdx.x;
    for (int idx = tid; idx < 4 * 64 * 64; idx += 256) {
        int khw = idx >> 12, q = (idx >> 6) & 63, ch = idx & 63;
        // squeeze: Ae[khw][q][ch] = w_sq2[q,khw] * w_sq1[q//2, ch]
        Ae[idx] = (_Float16)(w_sq2[q * 4 + khw] * w_sq1[(q >> 1) * 64 + ch]);
        // unsqueeze: We[s][o][2cc+p] = w_un2[o,cc] * w_un1[4cc+s, p]
        int s = khw, o = q, cc = ch >> 1, p = ch & 1;
        We[idx] = (_Float16)(w_un2[o * 32 + cc] * w_un1[(4 * cc + s) * 2 + p]);
    }
    if (tid < 64) {
        float sw = w_sq2[tid * 4] + w_sq2[tid * 4 + 1] + w_sq2[tid * 4 + 2] + w_sq2[tid * 4 + 3];
        Abias[tid] = b_sq2[tid] + b_sq1[tid >> 1] * sw;
    }
    if (tid < 256) {
        int s = tid >> 6, o = tid & 63;
        float acc = b_un2[o];
        for (int cc = 0; cc < 32; ++cc) acc += w_un2[o * 32 + cc] * b_un1[4 * cc + s];
        Wbias[tid] = acc;
    }
}

// ---------------------------------------------------------------------------
// K1: squeeze. z0[b,p,i,j] (p ca-shuffled) = sum_khw Ae_khw @ x_sub + Abias.
// Grid: B*16 blocks (8x8 output-pixel tiles), 256 threads (8 waves).
// 4 accumulated WMMA GEMMs: M=64(q), K=64(ch), N=64(px).
// ---------------------------------------------------------------------------
__global__ __launch_bounds__(256) void k_squeeze(
    const float* __restrict__ x, const _Float16* __restrict__ Ae,
    const float* __restrict__ Abias, float* __restrict__ z0) {
    __shared__ __align__(16) _Float16 Asrc[64 * 64];   // [q][ch]   8KB
    __shared__ __align__(16) _Float16 BsrcT[64 * 64];  // [px][ch]  8KB (operand-major)
    int b = blockIdx.x >> 4, tile = blockIdx.x & 15;
    int ti = (tile >> 2) * 8, tj = (tile & 3) * 8;
    int tid = threadIdx.x, lane = tid & 31, wid = tid >> 5;

    v8f acc[2];
    acc[0] = zero8(); acc[1] = zero8();

    for (int khw = 0; khw < 4; ++khw) {
        int kh = khw >> 1, kw = khw & 1;
        __syncthreads();
        {   // weight panel: contiguous f16 block -> 128-bit moves
            const uint4* src = (const uint4*)(Ae + khw * 4096);
            uint4* dst = (uint4*)Asrc;
#pragma unroll
            for (int idx = tid; idx < 512; idx += 256) dst[idx] = src[idx];
        }
        for (int idx = tid; idx < 4096; idx += 256) {
            int ch = idx >> 6, n = idx & 63;               // coalesced global read
            int pi = n >> 3, pj = n & 7;
            int ii = 2 * (ti + pi) + kh, jj = 2 * (tj + pj) + kw;
            BsrcT[n * 64 + ch] = (_Float16)x[((b * 64 + ch) * 64 + ii) * 64 + jj];
        }
        if (khw < 3)  // overlap next weight panel fetch with this GEMM stage
            __builtin_prefetch(Ae + (khw + 1) * 4096 + tid * 16, 0, 0);
        __syncthreads();
#pragma unroll
        for (int ks = 0; ks < 2; ++ks) {
#pragma unroll
            for (int p = 0; p < 2; ++p) {
                int id = wid * 2 + p;          // 16 output tiles: 4 mtiles x 4 ntiles
                int mt = id >> 2, nt = id & 3;
                v16h A = build_A16(Asrc, 64, mt, ks * 32, lane);
                v16h B = build_BT16(BsrcT, 64, nt, ks * 32, lane);
                acc[p] = wmma_f16(A, B, acc[p]);
            }
        }
    }
    int nlo = lane & 15, mh = (lane < 16) ? 0 : 8;
#pragma unroll
    for (int p = 0; p < 2; ++p) {
        int id = wid * 2 + p;
        int mt = id >> 2, nt = id & 3;
        int nfull = nt * 16 + nlo;
        int pi = nfull >> 3, pj = nfull & 7;
#pragma unroll
        for (int r = 0; r < 8; ++r) {
            int q  = mt * 16 + r + mh;
            int pc = 2 * (q & 31) + (q >> 5);  // ca channel shuffle folded in
            z0[(b * 64 + pc) * 1024 + (ti + pi) * 32 + (tj + pj)] = acc[p][r] + Abias[q];
        }
    }
}

// ---------------------------------------------------------------------------
// K2: channel attention (spatial=False). Heads m and m+32 read the same
// channel pair (2m,2m+1); attn is 2x2 per head -> reductions + elementwise.
// sa channel shuffle folded into the store. Grid: B*32, 256 threads.
// ---------------------------------------------------------------------------
__global__ __launch_bounds__(256) void k_chan_attn(
    const float* __restrict__ z0, const float* __restrict__ wqkv,
    const float* __restrict__ bqkv, const float* __restrict__ tca,
    const float* __restrict__ wf, const float* __restrict__ bf,
    float* __restrict__ z1) {
    __shared__ float wsum[8][16];
    __shared__ float sums[16];
    __shared__ float s_attn[8];  // [e][d][e']
    int b = blockIdx.x >> 5, m = blockIdx.x & 31;
    int tid = threadIdx.x, lane = tid & 31, wid = tid >> 5;
    int c0 = 2 * m, c1 = 2 * m + 1;
    const float* pa = z0 + (b * 64 + c0) * 1024;
    const float* pb = z0 + (b * 64 + c1) * 1024;

    // 16B-aligned vector loads: 4 consecutive pixels per thread
    float4 a4 = *(const float4*)(pa + tid * 4);
    float4 b4 = *(const float4*)(pb + tid * 4);
    float za[4] = {a4.x, a4.y, a4.z, a4.w};
    float zb[4] = {b4.x, b4.y, b4.z, b4.w};

    float part[16];
#pragma unroll
    for (int i = 0; i < 16; ++i) part[i] = 0.0f;
#pragma unroll
    for (int e = 0; e < 2; ++e) {
        float wq0 = wqkv[c0 * 6 + e],     bq0 = bqkv[c0 * 6 + e];
        float wq1 = wqkv[c1 * 6 + e],     bq1 = bqkv[c1 * 6 + e];
        float wk0 = wqkv[c0 * 6 + 2 + e], bk0 = bqkv[c0 * 6 + 2 + e];
        float wk1 = wqkv[c1 * 6 + 2 + e], bk1 = bqkv[c1 * 6 + 2 + e];
        float* p = part + e * 8;
#pragma unroll
        for (int u = 0; u < 4; ++u) {
            float q0 = za[u] * wq0 + bq0, q1 = zb[u] * wq1 + bq1;
            float k0 = za[u] * wk0 + bk0, k1 = zb[u] * wk1 + bk1;
            p[0] += q0 * q0; p[1] += q1 * q1; p[2] += k0 * k0; p[3] += k1 * k1;
            p[4] += q0 * k0; p[5] += q0 * k1; p[6] += q1 * k0; p[7] += q1 * k1;
        }
    }
    // deterministic reduction: wave shfl-tree -> per-wave partials -> 8-way sum
#pragma unroll
    for (int i = 0; i < 16; ++i) {
        float r = wred_sum32(part[i]);
        if (lane == 0) wsum[wid][i] = r;
    }
    __syncthreads();
    if (tid < 16) {
        float s = 0.0f;
        for (int w = 0; w < 8; ++w) s += wsum[w][tid];
        sums[tid] = s;
    }
    __syncthreads();
    if (tid < 2) {
        int e = tid;
        const float* p = sums + e * 8;
        float nq0 = fmaxf(sqrtf(p[0]), 1e-12f), nq1 = fmaxf(sqrtf(p[1]), 1e-12f);
        float nk0 = fmaxf(sqrtf(p[2]), 1e-12f), nk1 = fmaxf(sqrtf(p[3]), 1e-12f);
        float t = tca[32 * e + m];
        float s00 = t * p[4] / (nq0 * nk0), s01 = t * p[5] / (nq0 * nk1);
        float s10 = t * p[6] / (nq1 * nk0), s11 = t * p[7] / (nq1 * nk1);
        float m0 = fmaxf(s00, s01), m1 = fmaxf(s10, s11);
        float e00 = __expf(s00 - m0), e01 = __expf(s01 - m0);
        float e10 = __expf(s10 - m1), e11 = __expf(s11 - m1);
        float d0 = e00 + e01, d1 = e10 + e11;
        s_attn[e * 4 + 0] = e00 / d0; s_attn[e * 4 + 1] = e01 / d0;
        s_attn[e * 4 + 2] = e10 / d1; s_attn[e * 4 + 3] = e11 / d1;
    }
    __syncthreads();
    // out[c=2m+dd] = wf[c,0]*o_{e=0}[dd] + wf[c,1]*o_{e=1}[dd] + bf[c]
#pragma unroll
    for (int dd = 0; dd < 2; ++dd) {
        int c = 2 * m + dd;
        float A0 = wf[c * 2 + 0] * s_attn[0 * 4 + dd * 2 + 0];
        float A1 = wf[c * 2 + 0] * s_attn[0 * 4 + dd * 2 + 1];
        float B0 = wf[c * 2 + 1] * s_attn[1 * 4 + dd * 2 + 0];
        float B1 = wf[c * 2 + 1] * s_attn[1 * 4 + dd * 2 + 1];
        float bb = bf[c];
        int r = (dd == 0) ? m : (32 + m);  // sa channel shuffle folded in
        float* po = z1 + (b * 64 + r) * 1024;
        float wv00 = wqkv[c0 * 6 + 4 + 0], bv00 = bqkv[c0 * 6 + 4 + 0];
        float wv01 = wqkv[c1 * 6 + 4 + 0], bv01 = bqkv[c1 * 6 + 4 + 0];
        float wv10 = wqkv[c0 * 6 + 4 + 1], bv10 = bqkv[c0 * 6 + 4 + 1];
        float wv11 = wqkv[c1 * 6 + 4 + 1], bv11 = bqkv[c1 * 6 + 4 + 1];
        float4 o4;
        float* ov = (float*)&o4;
#pragma unroll
        for (int u = 0; u < 4; ++u) {
            float v00 = za[u] * wv00 + bv00, v01 = zb[u] * wv01 + bv01;
            float v10 = za[u] * wv10 + bv10, v11 = zb[u] * wv11 + bv11;
            ov[u] = A0 * v00 + A1 * v01 + B0 * v10 + B1 * v11 + bb;
        }
        *(float4*)(po + tid * 4) = o4;  // 16B-aligned vector store
    }
}

// ---------------------------------------------------------------------------
// K3: spatial attention, flash-style. One block per (b, m); computes both
// coupled heads (m, m+32), QK^T tiles via WMMA (d=2 zero-padded into K=32),
// online softmax in the C/D layout, group-fuse done in LDS.
// Grid: B*32, 256 threads (8 waves, 16 flash jobs each).
// ---------------------------------------------------------------------------
__global__ __launch_bounds__(256) void k_spatial_attn(
    const float* __restrict__ z1, const float* __restrict__ wqkv,
    const float* __restrict__ bqkv, const float* __restrict__ tsa,
    const float* __restrict__ wf, const float* __restrict__ bf,
    float* __restrict__ z2) {
    __shared__ _Float16 qsh[2][2][1024];  // [e][d][n], pre-scaled by t, l2-normed
    __shared__ _Float16 ksh[2][2][1024];  // l2-normed
    __shared__ _Float16 vsh[2][2][1024];
    __shared__ float    osh[2][2][1024];  // per-head outputs, fused at the end
    int b = blockIdx.x >> 5, m = blockIdx.x & 31;
    int tid = threadIdx.x, lane = tid & 31, wid = tid >> 5;
    int c0 = 2 * m, c1 = 2 * m + 1;
    const float* pa = z1 + (b * 64 + c0) * 1024;
    const float* pb = z1 + (b * 64 + c1) * 1024;

    {   // 16B-aligned vector preload: 4 consecutive pixels per thread
        int n0 = tid * 4;
        float4 a4 = *(const float4*)(pa + n0);
        float4 b4 = *(const float4*)(pb + n0);
        float za[4] = {a4.x, a4.y, a4.z, a4.w};
        float zb[4] = {b4.x, b4.y, b4.z, b4.w};
#pragma unroll
        for (int e = 0; e < 2; ++e) {
            float t   = tsa[32 * e + m];
            float wq0 = wqkv[c0 * 6 + e],     bq0 = bqkv[c0 * 6 + e];
            float wq1 = wqkv[c1 * 6 + e],     bq1 = bqkv[c1 * 6 + e];
            float wk0 = wqkv[c0 * 6 + 2 + e], bk0 = bqkv[c0 * 6 + 2 + e];
            float wk1 = wqkv[c1 * 6 + 2 + e], bk1 = bqkv[c1 * 6 + 2 + e];
            float wv0 = wqkv[c0 * 6 + 4 + e], bv0 = bqkv[c0 * 6 + 4 + e];
            float wv1 = wqkv[c1 * 6 + 4 + e], bv1 = bqkv[c1 * 6 + 4 + e];
#pragma unroll
            for (int u = 0; u < 4; ++u) {
                int n = n0 + u;
                float q0 = za[u] * wq0 + bq0, q1 = zb[u] * wq1 + bq1;
                float k0 = za[u] * wk0 + bk0, k1 = zb[u] * wk1 + bk1;
                float v0 = za[u] * wv0 + bv0, v1 = zb[u] * wv1 + bv1;
                float nq = fmaxf(sqrtf(q0 * q0 + q1 * q1), 1e-12f);
                float nk = fmaxf(sqrtf(k0 * k0 + k1 * k1), 1e-12f);
                qsh[e][0][n] = (_Float16)(q0 / nq * t);
                qsh[e][1][n] = (_Float16)(q1 / nq * t);
                ksh[e][0][n] = (_Float16)(k0 / nk);
                ksh[e][1][n] = (_Float16)(k1 / nk);
                vsh[e][0][n] = (_Float16)v0;
                vsh[e][1][n] = (_Float16)v1;
            }
        }
    }
    __syncthreads();

    int nlo = lane & 15;
    for (int jj = 0; jj < 16; ++jj) {
        int jb = wid * 16 + jj;          // 128 jobs = e(2) x qtile(64)
        int e = jb >> 6, qt = jb & 63;
        // A operand: rows = 16 queries, K: only K=0,1 live (held by lanes 0-15)
        v16h A = zero16h();
        if (lane < 16) {
            int qi = qt * 16 + lane;
            A[0] = qsh[e][0][qi];
            A[1] = qsh[e][1][qi];
        }
        float rowmax[8], rowsum[8], a0[8], a1[8];
#pragma unroll
        for (int r = 0; r < 8; ++r) { rowmax[r] = -1e30f; rowsum[r] = 0.0f; a0[r] = 0.0f; a1[r] = 0.0f; }

        for (int mt = 0; mt < 64; ++mt) {
            int ki = mt * 16 + nlo;
            v16h B = zero16h();
            if (lane < 16) {               // K=0,1 live (lanes>=16 hold K=8..15,24..31 pad)
                B[0] = ksh[e][0][ki];
                B[1] = ksh[e][1][ki];
            }
            v8f c = wmma_f16(A, B, zero8());  // S tile: lane holds rows r(+8), col nlo
            float v0 = (float)vsh[e][0][ki];
            float v1 = (float)vsh[e][1][ki];
#pragma unroll
            for (int r = 0; r < 8; ++r) {
                float s  = c[r];
                float tm = hred_max(s);
                float nm = fmaxf(rowmax[r], tm);
                float sc = __expf(rowmax[r] - nm);
                float p  = __expf(s - nm);
                float ps = hred_sum(p);
                rowsum[r] = rowsum[r] * sc + ps;
                a0[r] = a0[r] * sc + p * v0;   // per-lane partial of o[d]
                a1[r] = a1[r] * sc + p * v1;
                rowmax[r] = nm;
            }
        }
#pragma unroll
        for (int r = 0; r < 8; ++r) {
            float o0 = hred_sum(a0[r]) / rowsum[r];
            float o1 = hred_sum(a1[r]) / rowsum[r];
            int M  = r + ((lane < 16) ? 0 : 8);
            int qi = qt * 16 + M;
            if (nlo == 0) { osh[e][0][qi] = o0; osh[e][1][qi] = o1; }
        }
    }
    __syncthreads();
    // group fuse: out[c=2m+dd] = wf[c,0]*o_{head m}[dd] + wf[c,1]*o_{head m+32}[dd] + bf[c]
    for (int idx = tid; idx < 2048; idx += 256) {
        int dd = idx >> 10, n = idx & 1023;
        int c = 2 * m + dd;
        z2[(b * 64 + c) * 1024 + n] =
            wf[c * 2 + 0] * osh[0][dd][n] + wf[c * 2 + 1] * osh[1][dd][n] + bf[c];
    }
}

// ---------------------------------------------------------------------------
// K4: unsqueeze. out[b,o,2i+ri,2j+rj] = We_s @ z2 + Wbias_s (s = 2*ri+rj).
// Grid: B*16 blocks (8x8 pixel tiles of the 32x32 grid), 4 WMMA GEMMs.
// ---------------------------------------------------------------------------
__global__ __launch_bounds__(256) void k_unsqueeze(
    const float* __restrict__ z2, const _Float16* __restrict__ We,
    const float* __restrict__ Wbias, float* __restrict__ out) {
    __shared__ __align__(16) _Float16 Asrc[64 * 64];   // [o][cp]
    __shared__ __align__(16) _Float16 BsrcT[64 * 64];  // [px][cp] (operand-major)
    int b = blockIdx.x >> 4, tile = blockIdx.x & 15;
    int ti = (tile >> 2) * 8, tj = (tile & 3) * 8;
    int tid = threadIdx.x, lane = tid & 31, wid = tid >> 5;

    // z2 staging: float4 per thread (16B aligned: tj%8==0, pj0%4==0),
    // scatter 4 halves into the transposed operand tile.
    for (int idx = tid; idx < 1024; idx += 256) {
        int cp = idx >> 4, rem = idx & 15;
        int pi = rem >> 1, pj0 = (rem & 1) * 4;
        float4 v = *(const float4*)(z2 + (b * 64 + cp) * 1024 + (ti + pi) * 32 + (tj + pj0));
        int nbase = pi * 8 + pj0;
        BsrcT[(nbase + 0) * 64 + cp] = (_Float16)v.x;
        BsrcT[(nbase + 1) * 64 + cp] = (_Float16)v.y;
        BsrcT[(nbase + 2) * 64 + cp] = (_Float16)v.z;
        BsrcT[(nbase + 3) * 64 + cp] = (_Float16)v.w;
    }
    for (int s = 0; s < 4; ++s) {
        __syncthreads();
        {   // weight panel: contiguous f16 block -> 128-bit moves
            const uint4* src = (const uint4*)(We + s * 4096);
            uint4* dst = (uint4*)Asrc;
#pragma unroll
            for (int idx = tid; idx < 512; idx += 256) dst[idx] = src[idx];
        }
        if (s < 3)  // overlap next weight panel fetch with this GEMM stage
            __builtin_prefetch(We + (s + 1) * 4096 + tid * 16, 0, 0);
        __syncthreads();
        v8f acc[2];
        acc[0] = zero8(); acc[1] = zero8();
#pragma unroll
        for (int ks = 0; ks < 2; ++ks) {
#pragma unroll
            for (int p = 0; p < 2; ++p) {
                int id = wid * 2 + p;
                int mt = id >> 2, nt = id & 3;
                v16h A = build_A16(Asrc, 64, mt, ks * 32, lane);
                v16h B = build_BT16(BsrcT, 64, nt, ks * 32, lane);
                acc[p] = wmma_f16(A, B, acc[p]);
            }
        }
        int ri = s >> 1, rj = s & 1;
        int nlo = lane & 15, mh = (lane < 16) ? 0 : 8;
#pragma unroll
        for (int p = 0; p < 2; ++p) {
            int id = wid * 2 + p;
            int mt = id >> 2, nt = id & 3;
            int nfull = nt * 16 + nlo;
            int pi = nfull >> 3, pj = nfull & 7;
            int i = ti + pi, j = tj + pj;
#pragma unroll
            for (int r = 0; r < 8; ++r) {
                int o = mt * 16 + r + mh;
                // pixel shuffle folded into the store index
                out[((b * 64 + o) * 64 + 2 * i + ri) * 64 + (2 * j + rj)] = acc[p][r] + Wbias[s * 64 + o];
            }
        }
    }
}

// ---------------------------------------------------------------------------
extern "C" void kernel_launch(void* const* d_in, const int* in_sizes, int n_in,
                              void* d_out, int out_size, void* d_ws, size_t ws_size,
                              hipStream_t stream) {
    (void)in_sizes; (void)n_in; (void)out_size; (void)ws_size;
    const float* x       = (const float*)d_in[0];
    const float* w_sq1   = (const float*)d_in[1];
    const float* b_sq1   = (const float*)d_in[2];
    const float* w_sq2   = (const float*)d_in[3];
    const float* b_sq2   = (const float*)d_in[4];
    const float* ca_wqkv = (const float*)d_in[5];
    const float* ca_bqkv = (const float*)d_in[6];
    const float* ca_t    = (const float*)d_in[7];
    const float* ca_wf   = (const float*)d_in[8];
    const float* ca_bf   = (const float*)d_in[9];
    const float* sa_wqkv = (const float*)d_in[10];
    const float* sa_bqkv = (const float*)d_in[11];
    const float* sa_t    = (const float*)d_in[12];
    const float* sa_wf   = (const float*)d_in[13];
    const float* sa_bf   = (const float*)d_in[14];
    const float* w_un1   = (const float*)d_in[15];
    const float* b_un1   = (const float*)d_in[16];
    const float* w_un2   = (const float*)d_in[17];
    const float* b_un2   = (const float*)d_in[18];
    float* out = (float*)d_out;

    char* ws = (char*)d_ws;
    _Float16* Ae  = (_Float16*)(ws + 0);          // 4*64*64 f16 = 32KB
    _Float16* We  = (_Float16*)(ws + 32768);      // 32KB
    float* Abias  = (float*)(ws + 65536);         // 64 f32
    float* Wbias  = (float*)(ws + 65536 + 1024);  // 256 f32
    float* z0 = (float*)(ws + 131072);            // (2,64,32,32) f32 = 512KB
    float* z1 = (float*)(ws + 131072 + 524288);
    float* z2 = (float*)(ws + 131072 + 2 * 524288);

    k_prep<<<1, 256, 0, stream>>>(w_sq1, b_sq1, w_sq2, b_sq2,
                                  w_un1, b_un1, w_un2, b_un2,
                                  Ae, Abias, We, Wbias);
    k_squeeze<<<32, 256, 0, stream>>>(x, Ae, Abias, z0);
    k_chan_attn<<<64, 256, 0, stream>>>(z0, ca_wqkv, ca_bqkv, ca_t, ca_wf, ca_bf, z1);
    k_spatial_attn<<<64, 256, 0, stream>>>(z1, sa_wqkv, sa_bqkv, sa_t, sa_wf, sa_bf, z2);
    k_unsqueeze<<<32, 256, 0, stream>>>(z2, We, Wbias, out);
}
